// FeatureExtraction_91302414778573
// MI455X (gfx1250) — compile-verified, matrix operand order checked by reading, not run on hardware
//
#include <hip/hip_runtime.h>

typedef __attribute__((ext_vector_type(16))) _Float16 v16h;
typedef __attribute__((ext_vector_type(8)))  float    v8f;

#define B_   8
#define N_   4096
#define DIN  32
#define C0_  64
#define C1_  256
#define K_   16
#define NT   (B_ * N_)          // 32768 nodes
#define LRELU 0.2f

// ---------- helpers ----------

// Column index inside a 32-wide K-block for packed 16-bit WMMA operands:
// lanes 0-15 (half=0) hold K={0..7,16..23}, lanes 16-31 (half=1) hold K={8..15,24..31}
__device__ __forceinline__ int chunkc(int t, int half) {
  return (t < 8) ? (half * 8 + t) : (16 + half * 8 + (t - 8));
}

__device__ __forceinline__ v16h load16(const _Float16* row, int kb, int half) {
  v16h v;
#pragma unroll
  for (int t = 0; t < 16; ++t) v[t] = row[kb * 32 + chunkc(t, half)];
  return v;
}

// ---------- kernels ----------

__global__ void k_cvt_weights(const float* W0, const float* W1,
                              _Float16* W0h, _Float16* W1h) {
  int t = blockIdx.x * blockDim.x + threadIdx.x;
  if (t < C0_ * 2 * DIN)  W0h[t] = (_Float16)W0[t];
  if (t < C1_ * 2 * C0_)  W1h[t] = (_Float16)W1[t];
}

// y = x @ t0_W.T + t0_b ; emit f16 features + squared row norms (of rounded vals)
__global__ void k_transform0(const float* __restrict__ x, const float* __restrict__ W,
                             const float* __restrict__ bias,
                             _Float16* __restrict__ yh, float* __restrict__ sqv) {
  int wave = (blockIdx.x * blockDim.x + threadIdx.x) >> 5;
  int lane = threadIdx.x & 31;
  if (wave >= NT) return;
  float xv  = x[(size_t)wave * DIN + lane];
  float acc = bias[lane];
  const float* wrow = W + (size_t)lane * DIN;
#pragma unroll
  for (int k = 0; k < DIN; ++k) acc = fmaf(__shfl(xv, k), wrow[k], acc);
  _Float16 h = (_Float16)acc;
  yh[(size_t)wave * DIN + lane] = h;
  float s = (float)h * (float)h;
#pragma unroll
  for (int o = 16; o; o >>= 1) s += __shfl_xor(s, o);
  if (lane == 0) sqv[wave] = s;
}

// y1 = leaky_relu(h0 @ t1_W.T + t1_b) ; emit f16 features + squared row norms
__global__ void k_transform1(const float* __restrict__ h0, const float* __restrict__ W,
                             const float* __restrict__ bias,
                             _Float16* __restrict__ yh, float* __restrict__ sqv) {
  int wave = (blockIdx.x * blockDim.x + threadIdx.x) >> 5;
  int lane = threadIdx.x & 31;
  if (wave >= NT) return;
  const float* xr = h0 + (size_t)wave * C0_;
  float xa = xr[lane], xb = xr[lane + 32];
  float a0 = bias[lane], a1 = bias[lane + 32];
  const float* w0 = W + (size_t)lane * C0_;
  const float* w1 = W + (size_t)(lane + 32) * C0_;
#pragma unroll
  for (int k = 0; k < 32; ++k) {
    float xk = __shfl(xa, k);
    a0 = fmaf(xk, w0[k], a0);
    a1 = fmaf(xk, w1[k], a1);
  }
#pragma unroll
  for (int k = 0; k < 32; ++k) {
    float xk = __shfl(xb, k);
    a0 = fmaf(xk, w0[32 + k], a0);
    a1 = fmaf(xk, w1[32 + k], a1);
  }
  a0 = a0 > 0.f ? a0 : LRELU * a0;
  a1 = a1 > 0.f ? a1 : LRELU * a1;
  _Float16 ha = (_Float16)a0, hb = (_Float16)a1;
  yh[(size_t)wave * C0_ + lane]      = ha;
  yh[(size_t)wave * C0_ + 32 + lane] = hb;
  float s = (float)ha * (float)ha + (float)hb * (float)hb;
#pragma unroll
  for (int o = 16; o; o >>= 1) s += __shfl_xor(s, o);
  if (lane == 0) sqv[wave] = s;
}

// kNN within batch.
// Phase 1: 4 waves cooperatively fill a 16x4096 f32 d2 tile in LDS (bank-swizzled),
//          dot products via v_wmma_f32_16x16x32_f16 (A = j-tile, B = i-tile).
// Phase 2: each wave owns 4 rows; per row, 17 branchless wave-argmin extractions
//          (value, then index tie-break = stable top-k); rank 0 (self) dropped.
// Stores LOCAL j indices — faithful to the reference's local-index row bug.
template <int D>
__global__ void k_knn(const _Float16* __restrict__ feat, const float* __restrict__ sq,
                      int* __restrict__ knn) {
  constexpr int KB = D / 32;
  extern __shared__ float dsm[];            // 16 * 4096 floats = 256 KB
  const int bt = blockIdx.y;                // batch
  const int it = blockIdx.x;                // 16-row tile
  const int tid  = threadIdx.x;
  const int w    = tid >> 5;
  const int lane = tid & 31;
  const int lm   = lane & 15;
  const int half = lane >> 4;
  const int base = bt * N_;
  const int i0   = it * 16;

  // B operand: the i-tile (N dimension = i), constant for this workgroup
  v16h Bm[KB];
  {
    const _Float16* xi = feat + (size_t)(base + i0 + lm) * D;
#pragma unroll
    for (int kb = 0; kb < KB; ++kb) Bm[kb] = load16(xi, kb, half);
  }
  const float sqi = sq[base + i0 + lm];     // per-lane column (row i) norm
  const int   swzst = lm * 4;               // store-phase swizzle for this lane's row

  for (int jt = w; jt < N_ / 16; jt += 4) {
    const int j0 = jt * 16;
    const _Float16* xj = feat + (size_t)(base + j0 + lm) * D;
    v16h Am[KB];
#pragma unroll
    for (int kb = 0; kb < KB; ++kb) Am[kb] = load16(xj, kb, half);
    v8f c = {};
#pragma unroll
    for (int kb = 0; kb < KB; ++kb)
      c = __builtin_amdgcn_wmma_f32_16x16x32_f16(false, Am[kb], false, Bm[kb],
                                                 (short)0, c, false, false);
    // lane owns column N = lm (row i), rows M = j0 + r + 8*half across 8 VGPRs
#pragma unroll
    for (int r = 0; r < 8; ++r) {
      int j = j0 + r + 8 * half;
      float d2 = sqi + sq[base + j] - 2.0f * c[r];
      dsm[lm * N_ + (j ^ swzst)] = d2;
    }
  }
  __syncthreads();

  // Phase 2: wave w handles rows 4w .. 4w+3
  for (int rr = 0; rr < 4; ++rr) {
    const int i   = w * 4 + rr;
    float*    row = dsm + i * N_;
    const int swz = i * 4;
    for (int t = 0; t < 17; ++t) {
      float mn = __builtin_inff();
      int   mj = 0x7fffffff;
#pragma unroll 8
      for (int u = 0; u < N_ / 32; ++u) {
        int   j = lane + u * 32;
        float v = row[j ^ swz];
        bool  take = (v < mn);
        mn = take ? v : mn;
        mj = take ? j : mj;
      }
#pragma unroll
      for (int o = 16; o; o >>= 1) {
        float om = __shfl_xor(mn, o);
        int   oj = __shfl_xor(mj, o);
        bool  take = (om < mn) || (om == mn && oj < mj);
        mn = take ? om : mn;
        mj = take ? oj : mj;
      }
      if (lane == 0) {
        row[mj ^ swz] = __builtin_inff();   // invalidate winner
        if (t > 0) knn[(size_t)(base + i0 + i) * K_ + (t - 1)] = mj;
      }
    }
  }
}

// EdgeConv: per node (one wave), A = 16 edges x [x_i, x_j - x_i] (f16),
// WMMA against W tiles, per-edge bias+ReLU then max over edges (commuted)
template <int DI, int CO>
__global__ void k_edgeconv(const _Float16* __restrict__ feat, const int* __restrict__ knn,
                           const _Float16* __restrict__ Wh, const float* __restrict__ bias,
                           float* __restrict__ out) {
  constexpr int KB = (2 * DI) / 32;
  int wave = (blockIdx.x * blockDim.x + threadIdx.x) >> 5;
  int lane = threadIdx.x & 31;
  if (wave >= NT) return;
  const int i    = wave;
  const int lm   = lane & 15;
  const int half = lane >> 4;
  // NOTE: knn holds LOCAL per-batch indices used as GLOBAL rows (reference bug)
  const int j = knn[(size_t)i * K_ + lm];
  const _Float16* xi = feat + (size_t)i * DI;
  const _Float16* xj = feat + (size_t)j * DI;
  __builtin_prefetch(Wh, 0, 3);

  v16h A[KB];
#pragma unroll
  for (int kb = 0; kb < KB; ++kb) {
    v16h v;
#pragma unroll
    for (int t = 0; t < 16; ++t) {
      int c = kb * 32 + chunkc(t, half);
      v[t] = (c < DI) ? xi[c] : (_Float16)(xj[c - DI] - xi[c - DI]);
    }
    A[kb] = v;
  }

  for (int nt = 0; nt < CO / 16; ++nt) {
    const _Float16* wrow = Wh + (size_t)(nt * 16 + lm) * (2 * DI);
    v8f c = {};
#pragma unroll
    for (int kb = 0; kb < KB; ++kb) {
      v16h b = load16(wrow, kb, half);
      c = __builtin_amdgcn_wmma_f32_16x16x32_f16(false, A[kb], false, b,
                                                 (short)0, c, false, false);
    }
    float m = c[0];
#pragma unroll
    for (int r = 1; r < 8; ++r) m = fmaxf(m, c[r]);
    m = fmaxf(m, __shfl_xor(m, 16));               // combine M-halves
    m = fmaxf(m + bias[nt * 16 + lm], 0.0f);       // bias + ReLU after max
    if (half == 0) out[(size_t)i * CO + nt * 16 + lm] = m;
  }
}

// ---------- launch ----------

extern "C" void kernel_launch(void* const* d_in, const int* in_sizes, int n_in,
                              void* d_out, int out_size, void* d_ws, size_t ws_size,
                              hipStream_t stream) {
  const float* x    = (const float*)d_in[0];
  const float* t0W  = (const float*)d_in[1];
  const float* t0b  = (const float*)d_in[2];
  const float* W0   = (const float*)d_in[3];
  const float* b0   = (const float*)d_in[4];
  const float* t1W  = (const float*)d_in[5];
  const float* t1b  = (const float*)d_in[6];
  const float* W1   = (const float*)d_in[7];
  const float* b1   = (const float*)d_in[8];
  float* outp = (float*)d_out;

  // workspace carve-up (256B aligned)
  char* ws = (char*)d_ws;
  size_t off = 0;
  auto alloc = [&](size_t bytes) { void* p = ws + off; off = (off + bytes + 255) & ~(size_t)255; return p; };
  _Float16* y0h  = (_Float16*)alloc((size_t)NT * DIN * 2);
  float*    sq0  = (float*)   alloc((size_t)NT * 4);
  int*      knn1 = (int*)     alloc((size_t)NT * K_ * 4);
  float*    h0   = (float*)   alloc((size_t)NT * C0_ * 4);
  _Float16* y1h  = (_Float16*)alloc((size_t)NT * C0_ * 2);
  float*    sq1  = (float*)   alloc((size_t)NT * 4);
  int*      knn2 = (int*)     alloc((size_t)NT * K_ * 4);
  _Float16* W0h  = (_Float16*)alloc((size_t)C0_ * 2 * DIN * 2);
  _Float16* W1h  = (_Float16*)alloc((size_t)C1_ * 2 * C0_ * 2);
  (void)ws_size; (void)in_sizes; (void)n_in; (void)out_size;

  const int BLK = 128;                        // 4 waves
  const int waveBlocks = NT / 4;              // one wave per node/row
  const size_t knnLds = (size_t)16 * N_ * sizeof(float);   // 256 KB (fits 320 KB WGP LDS)

  k_cvt_weights<<<(C1_ * 2 * C0_ + 255) / 256, 256, 0, stream>>>(W0, W1, W0h, W1h);
  k_transform0<<<waveBlocks, BLK, 0, stream>>>(x, t0W, t0b, y0h, sq0);
  k_knn<DIN><<<dim3(N_ / 16, B_), BLK, knnLds, stream>>>(y0h, sq0, knn1);
  k_edgeconv<DIN, C0_><<<waveBlocks, BLK, 0, stream>>>(y0h, knn1, W0h, b0, h0);
  k_transform1<<<waveBlocks, BLK, 0, stream>>>(h0, t1W, t1b, y1h, sq1);
  k_knn<C0_><<<dim3(N_ / 16, B_), BLK, knnLds, stream>>>(y1h, sq1, knn2);
  k_edgeconv<C0_, C1_><<<waveBlocks, BLK, 0, stream>>>(y1h, knn2, W1h, b1, outp);
}
